// DNAclassificator_83099027243758
// MI455X (gfx1250) — matrix operand order repaired and checked by reading, not compiled, not run on hardware
//
#include <hip/hip_runtime.h>
#include <hip/hip_bf16.h>
#include <math.h>

// ---------------------------------------------------------------------------
// MI455X / gfx1250 implementation. Heavy GEMMs (MHA q/k/v + out projections,
// fc1) run on v_wmma_f32_16x16x32_f16 (f16 in, f32 accum). Convs / layernorm /
// mish / 8x8 softmax stay f32 VALU (bandwidth-bound, tiny channel counts).
// ---------------------------------------------------------------------------

typedef _Float16 f16;
typedef __attribute__((ext_vector_type(8)))  _Float16 v8h;
typedef __attribute__((ext_vector_type(16))) _Float16 v16h;
typedef __attribute__((ext_vector_type(8)))  float    v8f;

#define BATCH 4096

__device__ __forceinline__ float mishf(float x) {
    float sp = (x > 20.f) ? x : log1pf(expf(x));
    return x * tanhf(sp);
}

// ---------------------------------------------------------------------------
// WMMA GEMM:  C[m,n] = sum_k A[m,k] * W[n,k] + bias[n]   (A,W f16; C f32)
// Row m decodes as (b = m/8, s = m%8); addresses are
//   A row:  (b*a_cs + a_cg*g + s)*K        (g = blockIdx.z weight group)
//   W:      W + g*N*K + n*K
//   bias:   bias + g*N
//   C row:  (b*c_cs + c_cg*g + s)*N
// M % 64 == 0, N % 64 == 0, K % 32 == 0 (guaranteed by all call sites).
// 4 waves / block; wave w owns a 16x64 strip (4 v8f accumulators).
// A tile kept row-major in LDS (A frag = two contiguous 8-half runs/lane).
// W tile stored TRANSPOSED in LDS (Wst[k][n]); the B fragment (lane = N col,
// halves = 16 K rows, A-fragment order) is gathered from one Wst column ->
// 16 scalar u16 LDS reads per fragment.
// ---------------------------------------------------------------------------
__global__ void __launch_bounds__(128)
gemm_wmma_kernel(const f16* __restrict__ A, const f16* __restrict__ W,
                 const float* __restrict__ bias, float* __restrict__ C,
                 int M, int N, int K,
                 int a_cs, int a_cg, int c_cs, int c_cg, int domish)
{
    __shared__ f16 As [64][40];   // [m_local][k]   (80 B row stride, 16B-mult)
    __shared__ f16 Wst[32][72];   // [k][n_local]   (144 B row stride, 16B-mult)

    const int g    = blockIdx.z;
    const int n0   = blockIdx.x * 64;
    const int m0   = blockIdx.y * 64;
    const int tid  = threadIdx.x;
    const int wave = tid >> 5;
    const int lane = tid & 31;

    const f16*   Wg = W + (size_t)g * N * K;
    const float* bg = bias + (size_t)g * N;

    v8f acc[4];
    #pragma unroll
    for (int i = 0; i < 4; ++i)
        #pragma unroll
        for (int j = 0; j < 8; ++j) acc[i][j] = 0.f;

    for (int kk = 0; kk < K; kk += 32) {
        __syncthreads();
        // Stage A (row-major) and W (transposed) tiles; global reads are 16B.
        for (int task = tid; task < 256; task += 128) {
            int row = task >> 2;          // m_local / n_local
            int ch  = task & 3;           // 8-half k chunk
            int rg  = m0 + row;
            int bi  = rg >> 3, s = rg & 7;
            size_t aoff = ((size_t)bi * a_cs + a_cg * g + s) * K + kk + ch * 8;
            *(uint4*)(&As[row][ch * 8]) = *(const uint4*)(A + aoff);
            size_t woff = (size_t)(n0 + row) * K + kk + ch * 8;
            uint4 wv = *(const uint4*)(Wg + woff);
            const f16* wh = (const f16*)&wv;
            #pragma unroll
            for (int j = 0; j < 8; ++j)
                Wst[ch * 8 + j][row] = wh[j];   // transpose into [k][n]
            if (kk + 32 < K) {  // speculative prefetch of next K-slab
                __builtin_prefetch(A + aoff + 32, 0, 1);
                __builtin_prefetch(Wg + woff + 32, 0, 1);
            }
        }
        __syncthreads();

        // A fragment (ISA 7.12.2, 16-bit A 16x32):
        // lanes 0-15: M=lane, K in {0..7}U{16..23}; lanes 16-31: K in {8..15}U{24..31}
        const int arow = wave * 16 + (lane & 15);
        const int grp  = lane >> 4;
        v8h alo = *(const v8h*)&As[arow][grp ? 8 : 0];
        v8h ahi = *(const v8h*)&As[arow][grp ? 24 : 16];
        v16h af = __builtin_shufflevector(alo, ahi,
                    0, 1, 2, 3, 4, 5, 6, 7, 8, 9, 10, 11, 12, 13, 14, 15);

        // B fragment (32x16, K x N): N = lane%16, K in same order as A fragment.
        #pragma unroll
        for (int nt = 0; nt < 4; ++nt) {
            v16h bf;
            #pragma unroll
            for (int s = 0; s < 16; ++s)
                bf[s] = Wst[s < 8 ? grp * 8 + s : 16 + grp * 8 + (s - 8)][nt * 16 + (lane & 15)];
            acc[nt] = __builtin_amdgcn_wmma_f32_16x16x32_f16(
                false, af, false, bf, (short)0, acc[nt], false, false);
        }
    }

    // C layout: lane<16 -> (M=v, N=lane); lane>=16 -> (M=8+v, N=lane-16)
    int ncol = lane & 15;
    int mb   = (lane >> 4) * 8;
    #pragma unroll
    for (int nt = 0; nt < 4; ++nt) {
        int n    = n0 + nt * 16 + ncol;
        float bv = bg[n];
        #pragma unroll
        for (int v = 0; v < 8; ++v) {
            int m  = m0 + wave * 16 + mb + v;
            int bi = m >> 3, s = m & 7;
            float val = acc[nt][v] + bv;
            if (domish) val = mishf(val);
            C[((size_t)bi * c_cs + c_cg * g + s) * N + n] = val;
        }
    }
}

// ---------------------------------------------------------------------------
// Generic direct conv1d (f32). y layout (B, co_tot, Lout) with channel offset.
// ---------------------------------------------------------------------------
__global__ void conv1d_kernel(const float* __restrict__ x, const float* __restrict__ w,
                              const float* __restrict__ bias, float* __restrict__ y,
                              int Bt, int Cin, int Lin, int Cout, int Lout,
                              int co_off, int co_tot, int ktap, int padl, int stride,
                              int act, int accum)
{
    size_t idx = (size_t)blockIdx.x * blockDim.x + threadIdx.x;
    size_t total = (size_t)Bt * Cout * Lout;
    if (idx >= total) return;
    int t  = (int)(idx % Lout);
    int co = (int)((idx / Lout) % Cout);
    size_t b = idx / ((size_t)Lout * Cout);
    float acc = bias ? bias[co] : 0.f;
    const float* xb = x + b * (size_t)Cin * Lin;
    const float* wr = w + (size_t)co * Cin * ktap;
    for (int ci = 0; ci < Cin; ++ci)
        for (int j = 0; j < ktap; ++j) {
            int pos = t * stride + j - padl;
            if (pos >= 0 && pos < Lin)
                acc += wr[ci * ktap + j] * xb[(size_t)ci * Lin + pos];
        }
    if (act) acc = mishf(acc);
    float* yp = y + ((b * co_tot + co_off + co) * (size_t)Lout + t);
    if (accum) *yp += acc; else *yp = acc;
}

// conv 1x1, no bias, f16 output (qkv projection feed for WMMA)
__global__ void conv1x1_f16_kernel(const float* __restrict__ x, const float* __restrict__ w,
                                   f16* __restrict__ y, int Bt, int Cin, int L, int Cout)
{
    size_t idx = (size_t)blockIdx.x * blockDim.x + threadIdx.x;
    size_t total = (size_t)Bt * Cout * L;
    if (idx >= total) return;
    int t  = (int)(idx % L);
    int co = (int)((idx / L) % Cout);
    size_t b = idx / ((size_t)L * Cout);
    const float* xb = x + b * (size_t)Cin * L;
    const float* wr = w + (size_t)co * Cin;
    float acc = 0.f;
    for (int ci = 0; ci < Cin; ++ci)
        acc += wr[ci] * xb[(size_t)ci * L + t];
    y[(b * Cout + co) * (size_t)L + t] = (f16)acc;
}

// LayerNorm over the length axis (gamma/beta indexed by position), opt. mish.
__global__ void __launch_bounds__(256)
ln_kernel(const float* __restrict__ x, float* __restrict__ y,
          const float* __restrict__ gamma, const float* __restrict__ beta,
          int L, int domish)
{
    __shared__ float s1[256], s2[256];
    int tid = threadIdx.x;
    size_t row = blockIdx.x;
    const float* xr = x + row * (size_t)L;
    float a = 0.f, b = 0.f;
    for (int i = tid; i < L; i += 256) { float v = xr[i]; a += v; b += v * v; }
    s1[tid] = a; s2[tid] = b;
    __syncthreads();
    for (int st = 128; st > 0; st >>= 1) {
        if (tid < st) { s1[tid] += s1[tid + st]; s2[tid] += s2[tid + st]; }
        __syncthreads();
    }
    float mean = s1[0] / L;
    float var  = s2[0] / L - mean * mean;
    float rs   = rsqrtf(var + 1e-5f);
    float* yr = y + row * (size_t)L;
    for (int i = tid; i < L; i += 256) {
        float v = (xr[i] - mean) * rs * gamma[i] + beta[i];
        if (domish) v = mishf(v);
        yr[i] = v;
    }
}

// MHA core: S=8, H=8, Dh=E/8. P = (B,24,E) f32 (q ch0-7, k ch8-15, v ch16-23).
// O = (B,8,E) f32. One block per batch element; thread = (head, s).
__global__ void __launch_bounds__(64)
attn_kernel(const float* __restrict__ P, float* __restrict__ O, int E)
{
    __shared__ float sh[24 * 512];
    int b = blockIdx.x;
    const float* Pb = P + (size_t)b * 24 * E;
    for (int i = threadIdx.x; i < 24 * E; i += 64) sh[i] = Pb[i];
    __syncthreads();
    int h = threadIdx.x >> 3, s = threadIdx.x & 7;
    int Dh = E >> 3;
    float scale = 1.f / sqrtf((float)Dh);
    const float* q = sh + (size_t)s * E + h * Dh;
    float sc[8], mx = -1e30f;
    for (int t = 0; t < 8; ++t) {
        const float* k = sh + (size_t)(8 + t) * E + h * Dh;
        float d = 0.f;
        for (int dd = 0; dd < Dh; ++dd) d += q[dd] * k[dd];
        d *= scale; sc[t] = d; mx = fmaxf(mx, d);
    }
    float sum = 0.f;
    for (int t = 0; t < 8; ++t) { sc[t] = expf(sc[t] - mx); sum += sc[t]; }
    float inv = 1.f / sum;
    float* o = O + ((size_t)b * 8 + s) * E + h * Dh;
    for (int dd = 0; dd < Dh; ++dd) {
        float acc = 0.f;
        for (int t = 0; t < 8; ++t) acc += sc[t] * sh[(size_t)(16 + t) * E + h * Dh + dd];
        o[dd] = acc * inv;
    }
}

__global__ void cvt_f32_f16_kernel(const float* __restrict__ x, f16* __restrict__ y, size_t n) {
    size_t i = (size_t)blockIdx.x * blockDim.x + threadIdx.x;
    if (i < n) y[i] = (f16)x[i];
}
__global__ void add_kernel(float* __restrict__ y, const float* __restrict__ x, size_t n) {
    size_t i = (size_t)blockIdx.x * blockDim.x + threadIdx.x;
    if (i < n) y[i] += x[i];
}
__global__ void mish_kernel(float* __restrict__ y, size_t n) {
    size_t i = (size_t)blockIdx.x * blockDim.x + threadIdx.x;
    if (i < n) y[i] = mishf(y[i]);
}
__global__ void fc2_kernel(const float* __restrict__ u, const float* __restrict__ w2,
                           const float* __restrict__ b2, float* __restrict__ out, int Bt)
{
    int idx = blockIdx.x * blockDim.x + threadIdx.x;
    if (idx >= Bt * 2) return;
    int b = idx >> 1, j = idx & 1;
    const float* ur = u + (size_t)b * 64;
    float acc = b2[j];
    for (int k = 0; k < 64; ++k) acc += ur[k] * w2[j * 64 + k];
    out[idx] = acc;
}

// ---------------------------------------------------------------------------
// Host-side orchestration
// ---------------------------------------------------------------------------
static inline void conv(hipStream_t st, const float* x, const float* w, const float* bias,
                        float* y, int Bt, int Cin, int Lin, int Cout, int Lout,
                        int co_off, int co_tot, int k, int padl, int stride, int act, int accum)
{
    size_t total = (size_t)Bt * Cout * Lout;
    int blocks = (int)((total + 255) / 256);
    conv1d_kernel<<<blocks, 256, 0, st>>>(x, w, bias, y, Bt, Cin, Lin, Cout, Lout,
                                          co_off, co_tot, k, padl, stride, act, accum);
}
static inline void lnorm(hipStream_t st, const float* x, float* y, const float* g,
                         const float* b, int rows, int L, int domish)
{
    ln_kernel<<<rows, 256, 0, st>>>(x, y, g, b, L, domish);
}
static inline void cvt(hipStream_t st, const float* x, f16* y, size_t n)
{
    cvt_f32_f16_kernel<<<(int)((n + 255) / 256), 256, 0, st>>>(x, y, n);
}
static inline void gemm(hipStream_t st, const f16* A, const f16* W, const float* bias,
                        float* C, int M, int N, int K,
                        int a_cs, int a_cg, int c_cs, int c_cg, int G, int domish)
{
    dim3 grid(N / 64, M / 64, G);
    gemm_wmma_kernel<<<grid, 128, 0, st>>>(A, W, bias, C, M, N, K,
                                           a_cs, a_cg, c_cs, c_cg, domish);
}

struct BlkCfg { int nin, nout, dim, nsc, base; bool has_res; };

static void run_block(hipStream_t st, void* const* d_in, const BlkCfg& c,
                      float* X, float* A, float* Bb, f16* Cb, float* P, f16* Wh)
{
    auto F = [&](int off) { return (const float*)d_in[c.base + off]; };
    const int B = BATCH, E = c.dim, nsc = c.nsc, nin = c.nin, nout = c.nout;

    conv(st, X, F(0), F(1), A, B, nin, E, nsc, E, 0, nsc, 3, 1, 1, 0, 0);        // pre conv k3
    lnorm(st, A, Bb, F(2), F(3), B * nsc, E, 1);                                 // ln1 + mish
    conv(st, Bb, F(4), F(5), A, B, nsc, E, nsc, E, 0, nsc, 1, 0, 1, 0, 0);       // sc conv -> s3 in A
    lnorm(st, A, Bb, F(6), F(7), B * nsc, E, 0);                                 // attn ln -> xn in Bb
    {                                                                            // qkv conv (no bias) -> f16
        size_t total = (size_t)B * 24 * E;
        conv1x1_f16_kernel<<<(int)((total + 255) / 256), 256, 0, st>>>(Bb, F(8), Cb, B, nsc, E, 24);
    }
    cvt(st, F(9), Wh, (size_t)3 * E * E);                                        // in_w -> f16
    gemm(st, Cb, Wh, F(10), P, B * 8, E, E, 24, 8, 24, 8, 3, 0);                 // q/k/v projections (WMMA)
    attn_kernel<<<B, 64, 0, st>>>(P, (float*)Cb, E);                             // softmax attention -> o
    cvt(st, (const float*)Cb, (f16*)P, (size_t)B * 8 * E);                       // o -> f16
    cvt(st, F(11), Wh, (size_t)E * E);                                           // proj_w -> f16
    gemm(st, (f16*)P, Wh, F(12), (float*)Cb, B * 8, E, E, 8, 0, 8, 0, 1, 0);     // out projection (WMMA)
    conv(st, (const float*)Cb, F(13), F(14), A, B, 8, E, nsc, E, 0, nsc,
         1, 0, 1, 0, 1);                                                         // out conv, += s3 (in A)
    conv(st, A, F(15), F(16), Bb, B, nsc, E, nout, E, 0, nout, 3, 1, 1, 0, 0);   // post conv k3
    lnorm(st, Bb, Bb, F(17), F(18), B * nout, E, 1);                             // ln2 + mish (in place)
    if (c.has_res)
        conv(st, X, F(21), F(22), Bb, B, nin, E, nout, E, 0, nout, 1, 0, 1, 0, 1); // += res conv
    else
        add_kernel<<<(int)(((size_t)B * nout * E + 255) / 256), 256, 0, st>>>(Bb, X, (size_t)B * nout * E);
    conv(st, Bb, F(19), F(20), X, B, nout, E, nout, E / 2, 0, nout, 3, 1, 2, 0, 0); // down conv s2 -> X
}

extern "C" void kernel_launch(void* const* d_in, const int* in_sizes, int n_in,
                              void* d_out, int out_size, void* d_ws, size_t ws_size,
                              hipStream_t stream)
{
    (void)in_sizes; (void)n_in; (void)out_size; (void)ws_size;
    const int B = BATCH;
    const float* x_in = (const float*)d_in[0];
    auto In = [&](int i) { return (const float*)d_in[i]; };

    // Workspace pools (peak ~0.92 GB)
    char* w = (char*)d_ws;
    float* X  = (float*)w;  w += (size_t)B * 8192 * 4;    // block input / down output
    float* Ab = (float*)w;  w += (size_t)B * 12800 * 4;   // nsc*E max
    float* Bb = (float*)w;  w += (size_t)B * 16384 * 4;   // max(nsc,nout)*E max
    f16*   Cb = (f16*)w;    w += (size_t)B * 12288 * 2;   // qkv f16 / o f32 / proj-out f32
    float* P  = (float*)w;  w += (size_t)B * 12288 * 4;   // q/k/v projections, then o f16
    f16*   Wh = (f16*)w;    w += (size_t)2 * 1024 * 1024; // f16 weight staging

    // ---- 4-branch stem: conv k=1/2/3/4 (+mish), concat into X (B,16,512)
    conv(stream, x_in, In(1), In(2), X, B, 4, 512, 4, 512, 0,  16, 1, 0, 1, 1, 0);
    conv(stream, x_in, In(3), In(4), X, B, 4, 512, 4, 512, 4,  16, 2, 0, 1, 1, 0);
    conv(stream, x_in, In(5), In(6), X, B, 4, 512, 4, 512, 8,  16, 3, 1, 1, 1, 0);
    conv(stream, x_in, In(7), In(8), X, B, 4, 512, 4, 512, 12, 16, 4, 1, 1, 1, 0);

    // ---- 4 down blocks
    BlkCfg blks[4] = {
        {16, 32, 512, 25,  9, true },
        {32, 32, 256, 33, 32, false},
        {32, 16, 128, 25, 53, true },
        {16,  4,  64, 11, 76, true },
    };
    for (int i = 0; i < 4; ++i)
        run_block(stream, d_in, blks[i], X, Ab, Bb, Cb, P, Wh);

    // ---- head: mish, conv k3 (+mish), conv k1 (+mish), fc1 (WMMA +mish), fc2
    mish_kernel<<<(int)(((size_t)B * 128 + 255) / 256), 256, 0, stream>>>(X, (size_t)B * 128);
    conv(stream, X,  In(99),  In(100), Ab, B, 4, 32, 4, 32, 0, 4, 3, 1, 1, 1, 0);
    conv(stream, Ab, In(101), In(102), Bb, B, 4, 32, 2, 32, 0, 2, 1, 0, 1, 1, 0);
    cvt(stream, Bb, Cb, (size_t)B * 64);
    cvt(stream, In(103), Wh, 64 * 64);
    gemm(stream, Cb, Wh, In(104), Ab, B, 64, 64, 8, 0, 8, 0, 1, 1);
    fc2_kernel<<<(int)((B * 2 + 255) / 256), 256, 0, stream>>>(Ab, In(105), In(106), (float*)d_out, B);
}
